// ActorCriticRNN_7103875908086
// MI455X (gfx1250) — matrix operand order repaired and checked by reading, not compiled, not run on hardware
//
#include <hip/hip_runtime.h>
#include <hip/hip_bf16.h>
#include <math.h>

// ---------------------------------------------------------------------------
// Types for CDNA5 WMMA (wave32): D(8xf32) = A(16x32 f16) * B(32x16 f16) + C
// ---------------------------------------------------------------------------
typedef __attribute__((ext_vector_type(16))) _Float16 v16h;
typedef __attribute__((ext_vector_type(8)))  _Float16 v8h;
typedef __attribute__((ext_vector_type(8)))  float    v8f;

#define WMMA_F16(a, b, c) \
    __builtin_amdgcn_wmma_f32_16x16x32_f16(false, (a), false, (b), (short)0, (c), false, false)

// Problem constants
#define IN_SIZE 64
#define RNN 256
#define R3 768
#define BATCH 256
#define TT 1024
#define NA 16
#define MLPW 256

// ---------------------------------------------------------------------------
// Fragment loaders (A/B share a contiguous k order, not the ISA table order; C/D per CDNA5 ISA §7.12.2)
// A 16x32 f16: lane holds row m=lane&15, 16 contiguous K starting at k0+(lane>>4)*16
// B 32x16 f16: lane holds column n=lane&15, 16 contiguous K starting at k0+(lane>>4)*16
// C/D 16x16 f32: element v -> (m = v + 8*(lane>>4), n = lane&15)
// ---------------------------------------------------------------------------
__device__ inline v16h load_A_lds(const _Float16* base, int stride, int lane, int k0) {
    int m  = lane & 15;
    int hi = lane >> 4;
    const _Float16* p = base + m * stride + k0 + hi * 16;
    return *(const v16h*)p;
}

__device__ inline v16h load_B_glb(const _Float16* W, int K, int nbase, int k0, int lane) {
    int n  = nbase + (lane & 15);
    int ks = k0 + (lane >> 4) * 16;
    return *(const v16h*)(W + (size_t)n * K + ks);
}

__device__ inline float sigmoidf_(float x) { return 1.0f / (1.0f + __expf(-x)); }
__device__ inline float tanhf_(float x) {
    float ax = fabsf(x);
    float e  = __expf(-2.0f * ax);
    float r  = (1.0f - e) / (1.0f + e);
    return copysignf(r, x);
}

// ---------------------------------------------------------------------------
// Weight conversion f32 -> f16
// ---------------------------------------------------------------------------
__global__ void arc_cvt_f32f16(const float* __restrict__ s, _Float16* __restrict__ d, int n) {
    int i = blockIdx.x * 256 + threadIdx.x;
    if (i < n) d[i] = (_Float16)s[i];
}

// ---------------------------------------------------------------------------
// Phase 1: GRU scan. grid=16 WGs (16 batch rows each), block=512 (16 waves).
// Each wave owns ONE 16-column group (4 accumulators). K-loops partially
// unrolled so B-fragment staging stays bounded (VGPR budget = 256/wave at
// 4 waves/SIMD on a ~1024-entry register file).
// ---------------------------------------------------------------------------
__global__ __launch_bounds__(512, 1) void arc_gru_scan(
    const float* __restrict__ hidden, const float* __restrict__ obs,
    const unsigned char* __restrict__ done,
    const float* __restrict__ b_ih, const float* __restrict__ b_n,
    const _Float16* __restrict__ Wih, const _Float16* __restrict__ Whh,
    _Float16* __restrict__ hs, float* __restrict__ h_final)
{
    __shared__ __attribute__((aligned(32))) float    sh_h [16][RNN];      // 16 KB
    __shared__ __attribute__((aligned(32))) _Float16 sh_hh[16][RNN];      //  8 KB
    __shared__ __attribute__((aligned(32))) _Float16 sh_x [16][IN_SIZE];  //  2 KB
    __shared__ float sh_bih[R3];
    __shared__ float sh_bn [RNN];
    __shared__ unsigned char sh_done[16];

    const int tid  = threadIdx.x;
    const int bid  = blockIdx.x;          // 0..15, batch rows [bid*16, bid*16+16)
    const int wave = tid >> 5;            // 0..15 = column group
    const int lane = tid & 31;
    const int hi   = lane >> 4;
    const int ln   = lane & 15;
    const int g    = wave;
    const int col  = g * 16 + ln;

    for (int i = tid; i < R3; i += 512) sh_bih[i] = b_ih[i];
    if (tid < RNN) sh_bn[tid] = b_n[tid];
#pragma unroll
    for (int i = 0; i < 8; ++i) {
        int e = tid + 512 * i, m = e >> 8, c = e & 255;
        float v = hidden[(size_t)(bid * 16 + m) * RNN + c];
        sh_h[m][c]  = v;
        sh_hh[m][c] = (_Float16)v;
    }
    __syncthreads();

    for (int t = 0; t < TT; ++t) {
        // stage x_t (f16) and done flags
#pragma unroll
        for (int i = 0; i < 2; ++i) {
            int e = tid + 512 * i, m = e >> 6, k = e & 63;
            sh_x[m][k] = (_Float16)obs[((size_t)(bid * 16 + m) * TT + t) * IN_SIZE + k];
        }
        if (tid < 16) sh_done[tid] = done[(size_t)(bid * 16 + tid) * TT + t];
        __syncthreads();

        // episode reset: h = done ? 0 : h
#pragma unroll
        for (int i = 0; i < 8; ++i) {
            int e = tid + 512 * i, m = e >> 8, c = e & 255;
            if (sh_done[m]) { sh_h[m][c] = 0.0f; sh_hh[m][c] = (_Float16)0.0f; }
        }
        __syncthreads();

        // accumulators for this wave's 16 columns (bias-initialized)
        v8f r{}, z{}, in{}, hn{};
        {
            const float br = sh_bih[col], bz = sh_bih[RNN + col], bi = sh_bih[2 * RNN + col];
#pragma unroll
            for (int v = 0; v < 8; ++v) { r[v] = br; z[v] = bz; in[v] = bi; hn[v] = 0.0f; }
        }

        // x @ W_ih^T  (K = 64, 2 K-tiles)
#pragma unroll
        for (int k = 0; k < 2; ++k) {
            v16h a = load_A_lds(&sh_x[0][0], IN_SIZE, lane, k * 32);
            r  = WMMA_F16(a, load_B_glb(Wih, IN_SIZE,           g * 16, k * 32, lane), r);
            z  = WMMA_F16(a, load_B_glb(Wih, IN_SIZE, RNN     + g * 16, k * 32, lane), z);
            in = WMMA_F16(a, load_B_glb(Wih, IN_SIZE, 2 * RNN + g * 16, k * 32, lane), in);
        }
        // h @ W_hh^T  (K = 256, 8 K-tiles; bounded unroll -> bounded staging)
#pragma unroll 2
        for (int k = 0; k < 8; ++k) {
            v16h a = load_A_lds(&sh_hh[0][0], RNN, lane, k * 32);
            r  = WMMA_F16(a, load_B_glb(Whh, RNN,           g * 16, k * 32, lane), r);
            z  = WMMA_F16(a, load_B_glb(Whh, RNN, RNN     + g * 16, k * 32, lane), z);
            hn = WMMA_F16(a, load_B_glb(Whh, RNN, 2 * RNN + g * 16, k * 32, lane), hn);
        }
        __syncthreads();   // all LDS A-reads complete before h overwrite

        const bool last = (t == TT - 1);
        const float bnc = sh_bn[col];
#pragma unroll
        for (int v = 0; v < 8; ++v) {
            int m = v + 8 * hi;
            float rr = sigmoidf_(r[v]);
            float zz = sigmoidf_(z[v]);
            float nn = tanhf_(in[v] + rr * (hn[v] + bnc));
            float hold = sh_h[m][col];
            float hnew = nn + zz * (hold - nn);
            sh_h[m][col]  = hnew;
            _Float16 hh   = (_Float16)hnew;
            sh_hh[m][col] = hh;
            hs[((size_t)(bid * 16 + m) * TT + t) * RNN + col] = hh;
            if (last) h_final[(size_t)(bid * 16 + m) * RNN + col] = hnew;
        }
        __syncthreads();
    }
}

// ---------------------------------------------------------------------------
// Phase 2: policy/value MLP heads. 32 tokens per WG, 8 waves.
// ---------------------------------------------------------------------------
__device__ inline void layer256(const _Float16 (*src)[MLPW], _Float16 (*dst)[MLPW],
                                const _Float16* __restrict__ W, const float* __restrict__ bias,
                                bool relu, int wave, int lane)
{
    const int mtile = wave >> 2;       // 0..1
    const int hi = lane >> 4, ln = lane & 15;
    v16h a[8];
#pragma unroll
    for (int k = 0; k < 8; ++k) a[k] = load_A_lds(&src[mtile * 16][0], MLPW, lane, k * 32);
#pragma unroll
    for (int j = 0; j < 4; ++j) {
        const int ntile = (wave & 3) * 4 + j;
        const int col   = ntile * 16 + ln;
        const float bb  = bias[col];
        v8f c{};
#pragma unroll
        for (int v = 0; v < 8; ++v) c[v] = bb;
#pragma unroll 4
        for (int k = 0; k < 8; ++k)
            c = WMMA_F16(a[k], load_B_glb(W, MLPW, ntile * 16, k * 32, lane), c);
#pragma unroll
        for (int v = 0; v < 8; ++v) {
            float f = c[v];
            if (relu) f = fmaxf(f, 0.0f);
            dst[mtile * 16 + v + 8 * hi][col] = (_Float16)f;
        }
    }
}

__device__ inline void load_tile32(_Float16 (*dst)[MLPW], const _Float16* __restrict__ src, int tid) {
    const uint4* s = (const uint4*)src;
    uint4* d = (uint4*)&dst[0][0];
#pragma unroll
    for (int i = 0; i < 4; ++i) d[tid + 256 * i] = s[tid + 256 * i];
}

__global__ __launch_bounds__(256) void arc_mlp_heads(
    const _Float16* __restrict__ hs,
    const _Float16* __restrict__ W1p, const _Float16* __restrict__ W2p, const _Float16* __restrict__ W3p,
    const _Float16* __restrict__ W1v, const _Float16* __restrict__ W2v,
    const float* __restrict__ b1p, const float* __restrict__ b2p, const float* __restrict__ b3p,
    const float* __restrict__ b1v, const float* __restrict__ b2v,
    const float* __restrict__ W3v, const float* __restrict__ b3v,
    float* __restrict__ pol, float* __restrict__ val)
{
    __shared__ __attribute__((aligned(32))) _Float16 bufA[32][MLPW]; // 16 KB
    __shared__ __attribute__((aligned(32))) _Float16 bufB[32][MLPW]; // 16 KB

    const int tid  = threadIdx.x;
    const int wave = tid >> 5;
    const int lane = tid & 31;
    const int hi   = lane >> 4;
    const int ln   = lane & 15;
    const size_t tok0 = (size_t)blockIdx.x * 32;

    // ---- policy chain ----
    load_tile32(bufA, hs + tok0 * MLPW, tid);
    __syncthreads();
    layer256(bufA, bufB, W1p, b1p, true, wave, lane);
    __syncthreads();
    layer256(bufB, bufA, W2p, b2p, true, wave, lane);
    __syncthreads();

    // policy head: [32,256] @ [256,16]^T -> global (waves 0 and 4)
    if ((wave & 3) == 0) {
        const int mtile = wave >> 2;
        v16h a[8];
#pragma unroll
        for (int k = 0; k < 8; ++k) a[k] = load_A_lds(&bufA[mtile * 16][0], MLPW, lane, k * 32);
        const float bb = b3p[ln];
        v8f c{};
#pragma unroll
        for (int v = 0; v < 8; ++v) c[v] = bb;
#pragma unroll 4
        for (int k = 0; k < 8; ++k)
            c = WMMA_F16(a[k], load_B_glb(W3p, MLPW, 0, k * 32, lane), c);
#pragma unroll
        for (int v = 0; v < 8; ++v) {
            int m = mtile * 16 + v + 8 * hi;
            pol[(tok0 + m) * NA + ln] = c[v];
        }
    }
    // reload hs for the value chain (bufB is free)
    load_tile32(bufB, hs + tok0 * MLPW, tid);
    __syncthreads();

    // ---- value chain ----
    layer256(bufB, bufA, W1v, b1v, true, wave, lane);
    __syncthreads();
    layer256(bufA, bufB, W2v, b2v, true, wave, lane);
    __syncthreads();

    if (tid < 32) {
        float s = b3v[0];
#pragma unroll 8
        for (int k = 0; k < MLPW; ++k) s += (float)bufB[tid][k] * W3v[k];
        val[tok0 + tid] = s;
    }
}

// ---------------------------------------------------------------------------
// Host launcher
// ---------------------------------------------------------------------------
extern "C" void kernel_launch(void* const* d_in, const int* in_sizes, int n_in,
                              void* d_out, int out_size, void* d_ws, size_t ws_size,
                              hipStream_t stream) {
    const float* hidden = (const float*)d_in[0];
    const float* obs    = (const float*)d_in[1];
    const unsigned char* done = (const unsigned char*)d_in[2];
    const float* W_ih = (const float*)d_in[3];
    const float* W_hh = (const float*)d_in[4];
    const float* b_ih = (const float*)d_in[5];
    const float* b_n  = (const float*)d_in[6];
    const float* W1p = (const float*)d_in[7];  const float* b1p = (const float*)d_in[8];
    const float* W2p = (const float*)d_in[9];  const float* b2p = (const float*)d_in[10];
    const float* W3p = (const float*)d_in[11]; const float* b3p = (const float*)d_in[12];
    const float* W1v = (const float*)d_in[13]; const float* b1v = (const float*)d_in[14];
    const float* W2v = (const float*)d_in[15]; const float* b2v = (const float*)d_in[16];
    const float* W3v = (const float*)d_in[17]; const float* b3v = (const float*)d_in[18];

    char* ws = (char*)d_ws;
    _Float16* hWih = (_Float16*)(ws + 0);          // 768*64
    _Float16* hWhh = (_Float16*)(ws + 98304);      // 768*256
    _Float16* hW1p = (_Float16*)(ws + 491520);     // 256*256
    _Float16* hW2p = (_Float16*)(ws + 622592);
    _Float16* hW3p = (_Float16*)(ws + 753664);     // 16*256
    _Float16* hW1v = (_Float16*)(ws + 761856);
    _Float16* hW2v = (_Float16*)(ws + 892928);
    _Float16* hHs  = (_Float16*)(ws + 1024000);    // 256*1024*256 f16 = 128 MB

    auto cvt = [&](const float* s, _Float16* d, int n) {
        arc_cvt_f32f16<<<(n + 255) / 256, 256, 0, stream>>>(s, d, n);
    };
    cvt(W_ih, hWih, R3 * IN_SIZE);
    cvt(W_hh, hWhh, R3 * RNN);
    cvt(W1p,  hW1p, MLPW * MLPW);
    cvt(W2p,  hW2p, MLPW * MLPW);
    cvt(W3p,  hW3p, NA * MLPW);
    cvt(W1v,  hW1v, MLPW * MLPW);
    cvt(W2v,  hW2v, MLPW * MLPW);

    float* out     = (float*)d_out;
    float* h_final = out;                       // [256,256]
    float* pol     = out + BATCH * RNN;         // [256,1024,16]
    float* val     = pol + (size_t)BATCH * TT * NA; // [256,1024]

    arc_gru_scan<<<16, 512, 0, stream>>>(hidden, obs, done, b_ih, b_n, hWih, hWhh, hHs, h_final);

    arc_mlp_heads<<<(BATCH * TT) / 32, 256, 0, stream>>>(
        hHs, hW1p, hW2p, hW3p, hW1v, hW2v,
        b1p, b2p, b3p, b1v, b2v, W3v, b3v, pol, val);
}
